// TFSwinLayer_73942156968350
// MI455X (gfx1250) — matrix operand order repaired and checked by reading, not compiled, not run on hardware
//
#include <hip/hip_runtime.h>
#include <math.h>

// ---------------- problem constants ----------------
#define HH    56
#define CC    128
#define NTOK  49          // tokens per 7x7 window
#define BB    64          // batch
#define MTOK  (BB*HH*HH)  // 200704 total tokens
#define HEADS 4
#define HD    32
#define SHIFT3 3

typedef __attribute__((ext_vector_type(16))) __bf16 v16bf;
typedef __attribute__((ext_vector_type(8)))  float  v8f;

union Frag16 { v16bf v; uint4 q[2]; unsigned short h[16]; };

__device__ __forceinline__ unsigned short f2bf(float f) {
  unsigned u = __float_as_uint(f);
  u += 0x7fffu + ((u >> 16) & 1u);          // round-to-nearest-even
  return (unsigned short)(u >> 16);
}

__device__ __forceinline__ v8f wmma_bf16(const Frag16& a, const Frag16& b, v8f c) {
  return __builtin_amdgcn_wmma_f32_16x16x32_bf16(false, a.v, false, b.v,
                                                 (short)0, c, false, false);
}

// ---------------- weight transpose+convert: W[k][n] f32 -> Wt[n][k] bf16 ----
__global__ __launch_bounds__(256) void transpose_w(
    const float* __restrict__ src, unsigned short* __restrict__ dst, int K, int N)
{
  int idx = blockIdx.x * 256 + threadIdx.x;
  if (idx >= N * K) return;
  int n = idx / K, k = idx - n * K;
  dst[idx] = f2bf(src[(size_t)k * N + n]);
}

// ---------------- LN1 + cyclic shift + window partition -> xw bf16 ---------
__global__ __launch_bounds__(256) void ln1_shift_partition(
    const float* __restrict__ x, const float* __restrict__ g,
    const float* __restrict__ b, unsigned short* __restrict__ xw)
{
  int lane = threadIdx.x & 31;
  int t = blockIdx.x * 8 + (threadIdx.x >> 5);   // window-ordered token id
  int win = t / NTOK, n = t % NTOK;
  int img = win >> 6, wl = win & 63;
  int wh = wl >> 3, ww = wl & 7;
  int i = n / 7, j = n % 7;
  int sh = wh * 7 + i + SHIFT3; if (sh >= HH) sh -= HH;
  int sw = ww * 7 + j + SHIFT3; if (sw >= HH) sw -= HH;
  const float4* src = (const float4*)(x + ((size_t)img * (HH*HH) + sh*HH + sw) * CC) + lane;
  float4 xv = *src;
  float s = xv.x + xv.y + xv.z + xv.w;
#pragma unroll
  for (int m = 1; m < 32; m <<= 1) s += __shfl_xor(s, m);
  float mu = s * (1.f / CC);
  float dx = xv.x - mu, dy = xv.y - mu, dz = xv.z - mu, dw = xv.w - mu;
  float vs = dx*dx + dy*dy + dz*dz + dw*dw;
#pragma unroll
  for (int m = 1; m < 32; m <<= 1) vs += __shfl_xor(vs, m);
  float inv = rsqrtf(vs * (1.f / CC) + 1e-5f);
  float4 gv = ((const float4*)g)[lane];
  float4 bv = ((const float4*)b)[lane];
  ushort4 o;
  o.x = f2bf(dx * inv * gv.x + bv.x);
  o.y = f2bf(dy * inv * gv.y + bv.y);
  o.z = f2bf(dz * inv * gv.z + bv.z);
  o.w = f2bf(dw * inv * gv.w + bv.w);
  ((ushort4*)(xw + (size_t)t * CC))[lane] = o;
}

// ---------------- LN2 (natural token order) -> y bf16 ----------------------
__global__ __launch_bounds__(256) void ln_plain(
    const float* __restrict__ x, const float* __restrict__ g,
    const float* __restrict__ b, unsigned short* __restrict__ y)
{
  int lane = threadIdx.x & 31;
  int t = blockIdx.x * 8 + (threadIdx.x >> 5);
  const float4* src = (const float4*)(x + (size_t)t * CC) + lane;
  float4 xv = *src;
  float s = xv.x + xv.y + xv.z + xv.w;
#pragma unroll
  for (int m = 1; m < 32; m <<= 1) s += __shfl_xor(s, m);
  float mu = s * (1.f / CC);
  float dx = xv.x - mu, dy = xv.y - mu, dz = xv.z - mu, dw = xv.w - mu;
  float vs = dx*dx + dy*dy + dz*dz + dw*dw;
#pragma unroll
  for (int m = 1; m < 32; m <<= 1) vs += __shfl_xor(vs, m);
  float inv = rsqrtf(vs * (1.f / CC) + 1e-5f);
  float4 gv = ((const float4*)g)[lane];
  float4 bv = ((const float4*)b)[lane];
  ushort4 o;
  o.x = f2bf(dx * inv * gv.x + bv.x);
  o.y = f2bf(dy * inv * gv.y + bv.y);
  o.z = f2bf(dz * inv * gv.z + bv.z);
  o.w = f2bf(dw * inv * gv.w + bv.w);
  ((ushort4*)(y + (size_t)t * CC))[lane] = o;
}

// ---------------- generic WMMA GEMM: out = epi(A @ Bt^T + bias) ------------
// A [M,K] bf16 ; Bt [N,K] bf16 ; block tile 64(M) x 128(N), 8 waves.
// EPI 0: (acc+bias)*scale -> bf16 ; EPI 1: gelu(acc+bias) -> bf16
template <int EPI>
__global__ __launch_bounds__(256) void gemm_bf16(
    const unsigned short* __restrict__ A, const unsigned short* __restrict__ Bt,
    const float* __restrict__ bias, unsigned short* __restrict__ out,
    float scale, int N, int K)
{
  int lane = threadIdx.x & 31, w = threadIdx.x >> 5;
  int lmod = lane & 15, lhalf = lane >> 4;
  int rowbase = blockIdx.x * 64 + (w & 3) * 16;
  int colbase = blockIdx.y * 128 + (w >> 2) * 64;
  const unsigned short* arow = A + (size_t)(rowbase + lmod) * K + lhalf * 8;
  v8f acc[4];
#pragma unroll
  for (int tt = 0; tt < 4; ++tt) { v8f z = {}; acc[tt] = z; }
  for (int k0 = 0; k0 < K; k0 += 32) {
    Frag16 a;
    a.q[0] = *(const uint4*)(arow + k0);
    a.q[1] = *(const uint4*)(arow + k0 + 16);
#pragma unroll
    for (int tt = 0; tt < 4; ++tt) {
      const unsigned short* bp = Bt + (size_t)(colbase + tt*16 + lmod) * K + k0 + lhalf * 8;
      Frag16 bf;
      bf.q[0] = *(const uint4*)bp;
      bf.q[1] = *(const uint4*)(bp + 16);
      acc[tt] = wmma_bf16(a, bf, acc[tt]);
    }
  }
#pragma unroll
  for (int tt = 0; tt < 4; ++tt) {
    int col = colbase + tt * 16 + lmod;
    float bv = bias[col];
#pragma unroll
    for (int i = 0; i < 8; ++i) {
      int row = rowbase + i + lhalf * 8;
      float v = acc[tt][i] + bv;
      if (EPI == 0) v *= scale;
      else          v = 0.5f * v * (1.f + erff(v * 0.70710678118654752f));
      out[(size_t)row * N + col] = f2bf(v);
    }
  }
}

// ---------------- proj GEMM + window-reverse + unshift + residual ----------
__global__ __launch_bounds__(256) void gemm_proj(
    const unsigned short* __restrict__ A, const unsigned short* __restrict__ Bt,
    const float* __restrict__ bias, const float* __restrict__ shortcut,
    float* __restrict__ out)
{
  const int N = 128, K = 128;
  int lane = threadIdx.x & 31, w = threadIdx.x >> 5;
  int lmod = lane & 15, lhalf = lane >> 4;
  int rowbase = blockIdx.x * 64 + (w & 3) * 16;
  int colbase = (w >> 2) * 64;
  const unsigned short* arow = A + (size_t)(rowbase + lmod) * K + lhalf * 8;
  v8f acc[4];
#pragma unroll
  for (int tt = 0; tt < 4; ++tt) { v8f z = {}; acc[tt] = z; }
  for (int k0 = 0; k0 < K; k0 += 32) {
    Frag16 a;
    a.q[0] = *(const uint4*)(arow + k0);
    a.q[1] = *(const uint4*)(arow + k0 + 16);
#pragma unroll
    for (int tt = 0; tt < 4; ++tt) {
      const unsigned short* bp = Bt + (size_t)(colbase + tt*16 + lmod) * K + k0 + lhalf * 8;
      Frag16 bf;
      bf.q[0] = *(const uint4*)bp;
      bf.q[1] = *(const uint4*)(bp + 16);
      acc[tt] = wmma_bf16(a, bf, acc[tt]);
    }
  }
#pragma unroll
  for (int tt = 0; tt < 4; ++tt) {
    int col = colbase + tt * 16 + lmod;
    float bv = bias[col];
#pragma unroll
    for (int i = 0; i < 8; ++i) {
      int row = rowbase + i + lhalf * 8;       // window-ordered token id
      int win = row / NTOK, n = row - win * NTOK;
      int img = win >> 6, wl = win & 63, wh = wl >> 3, ww = wl & 7;
      int ii = n / 7, jj = n - ii * 7;
      int dh = wh * 7 + ii + SHIFT3; if (dh >= HH) dh -= HH;
      int dw = ww * 7 + jj + SHIFT3; if (dw >= HH) dw -= HH;
      size_t o = ((size_t)img * (HH*HH) + dh * HH + dw) * CC + col;
      out[o] = acc[tt][i] + bv + shortcut[o];
    }
  }
}

// ---------------- MLP2 GEMM + residual -> f32 out --------------------------
__global__ __launch_bounds__(256) void gemm_mlp2(
    const unsigned short* __restrict__ A, const unsigned short* __restrict__ Bt,
    const float* __restrict__ bias, const float* __restrict__ resid,
    float* __restrict__ out, int N, int K)
{
  int lane = threadIdx.x & 31, w = threadIdx.x >> 5;
  int lmod = lane & 15, lhalf = lane >> 4;
  int rowbase = blockIdx.x * 64 + (w & 3) * 16;
  int colbase = (w >> 2) * 64;
  const unsigned short* arow = A + (size_t)(rowbase + lmod) * K + lhalf * 8;
  v8f acc[4];
#pragma unroll
  for (int tt = 0; tt < 4; ++tt) { v8f z = {}; acc[tt] = z; }
  for (int k0 = 0; k0 < K; k0 += 32) {
    Frag16 a;
    a.q[0] = *(const uint4*)(arow + k0);
    a.q[1] = *(const uint4*)(arow + k0 + 16);
#pragma unroll
    for (int tt = 0; tt < 4; ++tt) {
      const unsigned short* bp = Bt + (size_t)(colbase + tt*16 + lmod) * K + k0 + lhalf * 8;
      Frag16 bf;
      bf.q[0] = *(const uint4*)bp;
      bf.q[1] = *(const uint4*)(bp + 16);
      acc[tt] = wmma_bf16(a, bf, acc[tt]);
    }
  }
#pragma unroll
  for (int tt = 0; tt < 4; ++tt) {
    int col = colbase + tt * 16 + lmod;
    float bv = bias[col];
#pragma unroll
    for (int i = 0; i < 8; ++i) {
      int row = rowbase + i + lhalf * 8;
      size_t o = (size_t)row * N + col;
      out[o] = acc[tt][i] + bv + resid[o];
    }
  }
}

// ---------------- per-window attention (one workgroup per window) ----------
__global__ __launch_bounds__(256) void attn_win(
    const unsigned short* __restrict__ qg, const unsigned short* __restrict__ kg,
    const unsigned short* __restrict__ vg, const float* __restrict__ bias_table,
    unsigned short* __restrict__ ctx)
{
  __shared__ unsigned short smem[3 * 64 * 128];    // q | k | vT ; probs overlay q|k
  __shared__ float sBias[169 * HEADS];
  unsigned short* sQ  = smem;
  unsigned short* sK  = smem + 64 * 128;
  unsigned short* sVT = smem + 2 * 64 * 128;       // [128][64] channel-major
  unsigned short* sP  = smem;                      // [4][64][64] probs (overlay)

  int tid = threadIdx.x;
  int lane = tid & 31, w = tid >> 5;
  int lmod = lane & 15, lhalf = lane >> 4;
  int win = blockIdx.x;
  int wl = win & 63, wh = wl >> 3, ww = wl & 7;
  size_t rowoff = (size_t)win * NTOK * CC;

  // stage q/k (vectorized) and vT (transposed) into LDS, zero-padded to 64 rows
  for (int idx = tid; idx < 1024; idx += 256) {
    int r = idx >> 4;
    int c8 = (idx & 15) * 8;
    uint4 qv = make_uint4(0, 0, 0, 0), kv = make_uint4(0, 0, 0, 0);
    if (r < NTOK) {
      qv = *(const uint4*)(qg + rowoff + (size_t)r * CC + c8);
      kv = *(const uint4*)(kg + rowoff + (size_t)r * CC + c8);
    }
    *(uint4*)(sQ + r * CC + c8) = qv;
    *(uint4*)(sK + r * CC + c8) = kv;
  }
  for (int idx = tid; idx < 64 * CC; idx += 256) {
    int r = idx >> 7, c = idx & 127;
    unsigned short val = (r < NTOK) ? vg[rowoff + (size_t)r * CC + c] : (unsigned short)0;
    sVT[c * 64 + r] = val;
  }
  for (int idx = tid; idx < 169 * HEADS; idx += 256) sBias[idx] = bias_table[idx];
  __syncthreads();

  // scores + softmax, all in registers. Wave owns 2 (head, mtile) row-blocks.
  v8f probs[2][4];
#pragma unroll
  for (int pp = 0; pp < 2; ++pp) {
    int pair = w * 2 + pp;
    int h = pair >> 2, mt = pair & 3;
    Frag16 a;
    const unsigned short* qrow = sQ + (mt * 16 + lmod) * CC + h * HD + lhalf * 8;
    a.q[0] = *(const uint4*)qrow;
    a.q[1] = *(const uint4*)(qrow + 16);
    v8f sc[4];
#pragma unroll
    for (int t = 0; t < 4; ++t) {
      Frag16 bf;
      const unsigned short* kp = sK + (t * 16 + lmod) * CC + h * HD + lhalf * 8;
      bf.q[0] = *(const uint4*)kp;
      bf.q[1] = *(const uint4*)(kp + 16);
      v8f z = {};
      sc[t] = wmma_bf16(a, bf, z);            // K=32 == head_dim: one WMMA/tile
    }
    // relative-position bias + shift mask + padding
#pragma unroll
    for (int t = 0; t < 4; ++t) {
      int ck = t * 16 + lmod;
#pragma unroll
      for (int i = 0; i < 8; ++i) {
        int rq = mt * 16 + i + lhalf * 8;
        float s;
        if (ck >= NTOK || rq >= NTOK) {
          s = -1e30f;
        } else {
          int iq = rq / 7, jq = rq - iq * 7;
          int ik = ck / 7, jk = ck - ik * 7;
          float bv = sBias[((iq - ik + 6) * 13 + (jq - jk + 6)) * HEADS + h];
          int hq = wh * 7 + iq, wq = ww * 7 + jq;
          int hk = wh * 7 + ik, wk = ww * 7 + jk;
          int rhq = hq < 49 ? 0 : (hq < 53 ? 1 : 2);
          int rwq = wq < 49 ? 0 : (wq < 53 ? 1 : 2);
          int rhk = hk < 49 ? 0 : (hk < 53 ? 1 : 2);
          int rwk = wk < 49 ? 0 : (wk < 53 ? 1 : 2);
          float mk = (rhq * 3 + rwq == rhk * 3 + rwk) ? 0.f : -100.f;
          s = sc[t][i] + bv + mk;
        }
        sc[t][i] = s;
      }
    }
    // row softmax: a row's 16 cols/tile live in one 16-lane half (xor 1/2/4/8)
#pragma unroll
    for (int i = 0; i < 8; ++i) {
      float mx = fmaxf(fmaxf(sc[0][i], sc[1][i]), fmaxf(sc[2][i], sc[3][i]));
#pragma unroll
      for (int m = 1; m < 16; m <<= 1) mx = fmaxf(mx, __shfl_xor(mx, m));
      float e0 = __expf(sc[0][i] - mx), e1 = __expf(sc[1][i] - mx);
      float e2 = __expf(sc[2][i] - mx), e3 = __expf(sc[3][i] - mx);
      float sum = e0 + e1 + e2 + e3;
#pragma unroll
      for (int m = 1; m < 16; m <<= 1) sum += __shfl_xor(sum, m);
      float inv = 1.f / sum;
      sc[0][i] = e0 * inv; sc[1][i] = e1 * inv; sc[2][i] = e2 * inv; sc[3][i] = e3 * inv;
    }
#pragma unroll
    for (int t = 0; t < 4; ++t) probs[pp][t] = sc[t];
  }
  __syncthreads();                      // q/k LDS reads done; safe to overlay probs
#pragma unroll
  for (int pp = 0; pp < 2; ++pp) {
    int pair = w * 2 + pp;
    int h = pair >> 2, mt = pair & 3;
#pragma unroll
    for (int t = 0; t < 4; ++t) {
      int col = t * 16 + lmod;
#pragma unroll
      for (int i = 0; i < 8; ++i) {
        int row = mt * 16 + i + lhalf * 8;
        sP[(h * 64 + row) * 64 + col] = f2bf(probs[pp][t][i]);
      }
    }
  }
  __syncthreads();

  // PV: ctx_h[64,32] = probs_h[64,64] @ v_h[64,32] ; 4 tiles per wave, K=64
#pragma unroll
  for (int s4 = 0; s4 < 4; ++s4) {
    int tidx = w * 4 + s4;
    int h = tidx >> 3, rem = tidx & 7, mt = rem >> 1, nt = rem & 1;
    v8f c = {};
#pragma unroll
    for (int ks = 0; ks < 2; ++ks) {
      Frag16 a;
      const unsigned short* pr = sP + (h * 64 + mt * 16 + lmod) * 64 + ks * 32 + lhalf * 8;
      a.q[0] = *(const uint4*)pr;
      a.q[1] = *(const uint4*)(pr + 16);
      Frag16 bf;
      const unsigned short* vp = sVT + (h * HD + nt * 16 + lmod) * 64 + ks * 32 + lhalf * 8;
      bf.q[0] = *(const uint4*)vp;
      bf.q[1] = *(const uint4*)(vp + 16);
      c = wmma_bf16(a, bf, c);
    }
#pragma unroll
    for (int i = 0; i < 8; ++i) {
      int row = mt * 16 + i + lhalf * 8;
      if (row < NTOK)
        ctx[rowoff + (size_t)row * CC + h * HD + nt * 16 + lmod] = f2bf(c[i]);
    }
  }
}

// ---------------- host orchestration ----------------------------------------
extern "C" void kernel_launch(void* const* d_in, const int* in_sizes, int n_in,
                              void* d_out, int out_size, void* d_ws, size_t ws_size,
                              hipStream_t stream)
{
  (void)in_sizes; (void)n_in; (void)out_size; (void)ws_size;
  const float* hidden = (const float*)d_in[0];
  const float* ln1_g  = (const float*)d_in[1];
  const float* ln1_b  = (const float*)d_in[2];
  const float* q_w    = (const float*)d_in[3];
  const float* q_b    = (const float*)d_in[4];
  const float* k_w    = (const float*)d_in[5];
  const float* k_b    = (const float*)d_in[6];
  const float* v_w    = (const float*)d_in[7];
  const float* v_b    = (const float*)d_in[8];
  const float* relb   = (const float*)d_in[9];
  const float* proj_w = (const float*)d_in[10];
  const float* proj_b = (const float*)d_in[11];
  const float* ln2_g  = (const float*)d_in[12];
  const float* ln2_b  = (const float*)d_in[13];
  const float* m1_w   = (const float*)d_in[14];
  const float* m1_b   = (const float*)d_in[15];
  const float* m2_w   = (const float*)d_in[16];
  const float* m2_b   = (const float*)d_in[17];

  char* ws = (char*)d_ws;
  const size_t SZ1 = (size_t)MTOK * CC * 2;                 // bf16 [M,128] buffer
  unsigned short* WqT = (unsigned short*)(ws + 0);          // [128][128] bf16
  unsigned short* WkT = (unsigned short*)(ws + 32768);
  unsigned short* WvT = (unsigned short*)(ws + 65536);
  unsigned short* WpT = (unsigned short*)(ws + 98304);
  unsigned short* W1T = (unsigned short*)(ws + 131072);     // [512][128] bf16
  unsigned short* W2T = (unsigned short*)(ws + 262144);     // [128][512] bf16
  const size_t base = 524288;
  unsigned short* XW = (unsigned short*)(ws + base);            // ln1 out (win order)
  unsigned short* Q  = (unsigned short*)(ws + base + 1 * SZ1);
  unsigned short* Kb = (unsigned short*)(ws + base + 2 * SZ1);
  unsigned short* Vb = (unsigned short*)(ws + base + 3 * SZ1);
  float*          Xf = (float*)(ws + base + 4 * SZ1);           // f32 residual, 2*SZ1 B
  unsigned short* H1 = (unsigned short*)(ws + base + 6 * SZ1);  // [M,512] bf16, 4*SZ1 B
  unsigned short* CTX = Q;   // safe alias: attn reads its window's q into LDS first
  unsigned short* Y   = XW;  // safe alias: XW dead after QKV GEMMs

  transpose_w<<<64,  256, 0, stream>>>(q_w,    WqT, 128, 128);
  transpose_w<<<64,  256, 0, stream>>>(k_w,    WkT, 128, 128);
  transpose_w<<<64,  256, 0, stream>>>(v_w,    WvT, 128, 128);
  transpose_w<<<64,  256, 0, stream>>>(proj_w, WpT, 128, 128);
  transpose_w<<<256, 256, 0, stream>>>(m1_w,   W1T, 128, 512);
  transpose_w<<<256, 256, 0, stream>>>(m2_w,   W2T, 512, 128);

  ln1_shift_partition<<<MTOK / 8, 256, 0, stream>>>(hidden, ln1_g, ln1_b, XW);

  dim3 g1(MTOK / 64, 1);
  gemm_bf16<0><<<g1, 256, 0, stream>>>(XW, WqT, q_b, Q,  0.17677669529663687f, 128, 128);
  gemm_bf16<0><<<g1, 256, 0, stream>>>(XW, WkT, k_b, Kb, 1.0f, 128, 128);
  gemm_bf16<0><<<g1, 256, 0, stream>>>(XW, WvT, v_b, Vb, 1.0f, 128, 128);

  attn_win<<<BB * 64, 256, 0, stream>>>(Q, Kb, Vb, relb, CTX);

  gemm_proj<<<g1, 256, 0, stream>>>(CTX, WpT, proj_b, hidden, Xf);

  ln_plain<<<MTOK / 8, 256, 0, stream>>>(Xf, ln2_g, ln2_b, Y);

  dim3 g2(MTOK / 64, 4);
  gemm_bf16<1><<<g2, 256, 0, stream>>>(Y, W1T, m1_b, H1, 1.0f, 512, 128);

  gemm_mlp2<<<g1, 256, 0, stream>>>(H1, W2T, m2_b, Xf, (float*)d_out, 128, 512);
}
